// GCN_54460185313830
// MI455X (gfx1250) — matrix operand order repaired and checked by reading, not compiled, not run on hardware
//
#include <hip/hip_runtime.h>
#include <hip/hip_bf16.h>

// ---------------------------------------------------------------------------
// GCN forward for MI455X (gfx1250, wave32, WMMA).
//   h1  = relu( scatter_add( (x@W1)[src] * ew -> dst ) + b1 )
//   out = log_softmax( scatter_add( (h1@W2)[src] * ew -> dst ) + b2 )
// GEMMs: v_wmma_f32_16x16x32_bf16. A and B operands are pre-swizzled into the
// exact ISA fragment layouts so the inner loop is b128 loads + wmma only.
// ---------------------------------------------------------------------------

typedef __attribute__((ext_vector_type(16))) __bf16 v16bf;
typedef __attribute__((ext_vector_type(8)))  float  v8f;

constexpr int NODES = 50000;   // 3125 * 16 exactly
constexpr int NF    = 512;     // K of GEMM1  (16 k-steps of 32)
constexpr int NH    = 128;     // N of GEMM1 / K of GEMM2 (4 k-steps)
constexpr int NC    = 40;      // N of GEMM2 (padded to 48 = 3 tiles)

static __device__ __forceinline__ unsigned pack_bf16(float lo, float hi) {
    unsigned short a = __builtin_bit_cast(unsigned short, (__bf16)lo);
    unsigned short b = __builtin_bit_cast(unsigned short, (__bf16)hi);
    return (unsigned)a | ((unsigned)b << 16);
}

static __device__ __forceinline__ v8f wmma_bf16(v16bf a, v16bf b, v8f c) {
    return __builtin_amdgcn_wmma_f32_16x16x32_bf16(
               false, a, false, b, (short)0, c, false, false);
}

// ---------------------------------------------------------------------------
// One-time weight pre-swizzle into B-fragment order:
//   B frag, lane L: col N = L%16, k = kstep*32 + (t/8)*16 + (L/16)*8 + t%8.
//   Layout: [(ntile*KSTEPS + kstep)*32 + lane]*16 + t   (t = 0..15)
// ---------------------------------------------------------------------------
__global__ void __launch_bounds__(256) gcn_cvt_w1(const float* __restrict__ W1,
                                                  __bf16* __restrict__ W1B) {
    const int p = blockIdx.x * 256 + threadIdx.x;        // 0 .. 8*16*32*16-1
    const int t    = p & 15;
    const int lane = (p >> 4) & 31;
    const int ks   = (p >> 9) & 15;
    const int nt   = p >> 13;
    const int ncol = nt * 16 + (lane & 15);
    const int k    = ks * 32 + ((t >> 3) << 4) + ((lane >> 4) << 3) + (t & 7);
    W1B[p] = (__bf16)W1[k * NH + ncol];
}

__global__ void __launch_bounds__(256) gcn_cvt_w2(const float* __restrict__ W2,
                                                  __bf16* __restrict__ W2B) {
    const int p = blockIdx.x * 256 + threadIdx.x;        // 0 .. 3*4*32*16-1
    if (p >= 3 * 4 * 32 * 16) return;
    const int t    = p & 15;
    const int lane = (p >> 4) & 31;
    const int ks   = (p >> 9) & 3;
    const int nt   = p >> 11;
    const int ncol = nt * 16 + (lane & 15);
    const int k    = ks * 32 + ((t >> 3) << 4) + ((lane >> 4) << 3) + (t & 7);
    W2B[p] = (ncol < NC) ? (__bf16)W2[k * NC + ncol] : (__bf16)0.0f;
}

// ---------------------------------------------------------------------------
// GEMM1: H[50000,128] = X[50000,512] @ W1[512,128]
// Block = 8 waves = one 16-row M panel; wave w owns N-tile w.
// A panel staged in LDS pre-swizzled to A-fragment order (u32 = bf16 pair):
//   u32 index q = kstep*256 + lane*8 + ii ; covers k = ks*32 + (ii/4)*16
//                 + (lane/16)*8 + (ii%4)*2 (+1), row = lane%16.
// Inner loop: 2x ds_load_b128 + 2x global_load_b128 + v_wmma.
// ---------------------------------------------------------------------------
__global__ void __launch_bounds__(256) gcn_gemm1(const float* __restrict__ X,
                                                 const __bf16* __restrict__ W1B,
                                                 float* __restrict__ H) {
    __shared__ unsigned As[16 * 256];                    // 16 KB swizzled A panel
    const int mtile = blockIdx.x;

    for (int q = threadIdx.x; q < 16 * 256; q += 256) {
        const int ks   = q >> 8;
        const int lane = (q >> 3) & 31;
        const int ii   = q & 7;
        const int r    = lane & 15;
        const int half = lane >> 4;
        const int k    = ks * 32 + ((ii >> 2) << 4) + (half << 3) + ((ii & 3) << 1);
        const float* xp = X + (size_t)(mtile * 16 + r) * NF + k;
        As[q] = pack_bf16(xp[0], xp[1]);
    }
    __syncthreads();

    const int wave = threadIdx.x >> 5;                   // ntile 0..7
    const int lane = threadIdx.x & 31;
    const int r    = lane & 15;
    const int half = lane >> 4;
    const int ncol = wave * 16 + r;
    const v16bf* bfrags = (const v16bf*)W1B + (size_t)wave * 16 * 32 + lane;

    v8f acc = {};
#pragma unroll 4
    for (int ks = 0; ks < 16; ++ks) {
        const v16bf a = *(const v16bf*)(&As[ks * 256 + lane * 8]);
        const v16bf b = bfrags[ks * 32];
        acc = wmma_bf16(a, b, acc);
    }

#pragma unroll
    for (int v = 0; v < 8; ++v) {                        // D: row = half*8+v
        const int m = mtile * 16 + half * 8 + v;
        H[(size_t)m * NH + ncol] = acc[v];
    }
}

// ---------------------------------------------------------------------------
// GEMM2: T2[50000,40] = relu(AG1)[50000,128] @ W2[128,40]  (N padded to 48)
// Activations arrive pre-swizzled in A-fragment order (HbfSw), so the loop is
// pure b128 loads + wmma. 3 waves / block, one 16-wide N-tile each.
// ---------------------------------------------------------------------------
__global__ void __launch_bounds__(96) gcn_gemm2(const unsigned* __restrict__ HbfSw,
                                                const __bf16* __restrict__ W2B,
                                                float* __restrict__ T2) {
    const int mtile = blockIdx.x;
    const int wave  = threadIdx.x >> 5;                  // ntile 0..2
    const int lane  = threadIdx.x & 31;
    const int r     = lane & 15;
    const int half  = lane >> 4;
    const int ncol  = wave * 16 + r;

    const v16bf* afrags = (const v16bf*)(HbfSw + (size_t)mtile * 1024) + lane;
    const v16bf* bfrags = (const v16bf*)W2B + (size_t)wave * 4 * 32 + lane;

    v8f acc = {};
#pragma unroll
    for (int ks = 0; ks < 4; ++ks) {
        acc = wmma_bf16(afrags[ks * 32], bfrags[ks * 32], acc);
    }

    if (ncol < NC) {
#pragma unroll
        for (int v = 0; v < 8; ++v) {
            const int m = mtile * 16 + half * 8 + v;
            T2[(size_t)m * NC + ncol] = acc[v];
        }
    }
}

// ---------------------------------------------------------------------------
// Edge scatter: one wave per edge (HBM/atomic roofline term).
// ---------------------------------------------------------------------------
__global__ void __launch_bounds__(256) gcn_scatter1(const float* __restrict__ H,
                                                    const int* __restrict__ src,
                                                    const int* __restrict__ dst,
                                                    const float* __restrict__ ew,
                                                    float* __restrict__ agg, int E) {
    const int e = blockIdx.x * 8 + (threadIdx.x >> 5);
    if (e >= E) return;
    const int   lane = threadIdx.x & 31;
    const int   s = src[e], d = dst[e];
    const float w = ew[e];
    const float4 v = *reinterpret_cast<const float4*>(H + (size_t)s * NH + lane * 4);
    float* o = agg + (size_t)d * NH + lane * 4;
    atomicAdd(o + 0, v.x * w);
    atomicAdd(o + 1, v.y * w);
    atomicAdd(o + 2, v.z * w);
    atomicAdd(o + 3, v.w * w);
}

__global__ void __launch_bounds__(256) gcn_scatter2(const float* __restrict__ T2,
                                                    const int* __restrict__ src,
                                                    const int* __restrict__ dst,
                                                    const float* __restrict__ ew,
                                                    float* __restrict__ out, int E) {
    const int e = blockIdx.x * 8 + (threadIdx.x >> 5);
    if (e >= E) return;
    const int   lane = threadIdx.x & 31;
    const int   s = src[e], d = dst[e];
    const float w = ew[e];
    if (lane < NC)
        atomicAdd(out + (size_t)d * NC + lane, T2[(size_t)s * NC + lane] * w);
    const int c2 = lane + 32;
    if (c2 < NC)
        atomicAdd(out + (size_t)d * NC + c2, T2[(size_t)s * NC + c2] * w);
}

// ---------------------------------------------------------------------------
// Epilogues
// ---------------------------------------------------------------------------
__global__ void __launch_bounds__(256) gcn_bias_init(float* __restrict__ o,
                                                     const float* __restrict__ b,
                                                     int total, int cols) {
    const int i = blockIdx.x * blockDim.x + threadIdx.x;
    if (i < total) o[i] = b[i % cols];
}

// ReLU + bf16 convert + scatter into A-fragment order for GEMM2.
// Thread i handles the k-pair (k, k+1) of node m -> one u32 store.
__global__ void __launch_bounds__(256) gcn_relu_swz(const float* __restrict__ agg,
                                                    unsigned* __restrict__ HbfSw) {
    const int i = blockIdx.x * blockDim.x + threadIdx.x;
    if (i >= NODES * (NH / 2)) return;
    const int m  = i >> 6;
    const int kp = (i & 63) << 1;
    const float2 v = *reinterpret_cast<const float2*>(agg + (size_t)m * NH + kp);
    const float lo = fmaxf(v.x, 0.0f), hi = fmaxf(v.y, 0.0f);
    const int mtile = m >> 4, r = m & 15;
    const int ks = kp >> 5, kq = kp & 31;
    const int half = (kq >> 3) & 1;
    const int lane = r + (half << 4);
    const int ii   = ((kq >> 4) << 2) + ((kq & 7) >> 1);
    HbfSw[(size_t)mtile * 1024 + ks * 256 + lane * 8 + ii] = pack_bf16(lo, hi);
}

__global__ void __launch_bounds__(256) gcn_log_softmax(float* __restrict__ o,
                                                       int nodes) {
    const int n = blockIdx.x * blockDim.x + threadIdx.x;
    if (n >= nodes) return;
    float v[NC];
    float mx = -3.402823e38f;
#pragma unroll
    for (int c = 0; c < NC; ++c) { v[c] = o[(size_t)n * NC + c]; mx = fmaxf(mx, v[c]); }
    float s = 0.0f;
#pragma unroll
    for (int c = 0; c < NC; ++c) s += __expf(v[c] - mx);
    const float ls = mx + __logf(s);
#pragma unroll
    for (int c = 0; c < NC; ++c) o[(size_t)n * NC + c] = v[c] - ls;
}

// ---------------------------------------------------------------------------
extern "C" void kernel_launch(void* const* d_in, const int* in_sizes, int n_in,
                              void* d_out, int out_size, void* d_ws, size_t ws_size,
                              hipStream_t stream) {
    (void)n_in; (void)out_size; (void)ws_size;
    const float* x   = (const float*)d_in[0];
    const int*   eix = (const int*)  d_in[1];
    const float* ew  = (const float*)d_in[2];
    const float* W1  = (const float*)d_in[3];
    const float* b1  = (const float*)d_in[4];
    const float* W2  = (const float*)d_in[5];
    const float* b2  = (const float*)d_in[6];
    float*       out = (float*)d_out;

    const int E = in_sizes[2];
    const int* src = eix;
    const int* dst = eix + E;

    char*  ws = (char*)d_ws;
    size_t off = 0;
    auto carve = [&](size_t bytes) -> void* {
        void* p = ws + off;
        off += (bytes + 255) & ~(size_t)255;
        return p;
    };
    float*    H1   = (float*)   carve(sizeof(float)    * (size_t)NODES * NH);
    float*    AG1  = (float*)   carve(sizeof(float)    * (size_t)NODES * NH);
    unsigned* HBSW = (unsigned*)carve(sizeof(unsigned) * (size_t)NODES * (NH / 2));
    float*    T2   = (float*)   carve(sizeof(float)    * (size_t)NODES * NC);
    __bf16*   W1B  = (__bf16*)  carve(sizeof(__bf16)   * 8 * 16 * 32 * 16);
    __bf16*   W2B  = (__bf16*)  carve(sizeof(__bf16)   * 3 * 4 * 32 * 16);

    gcn_cvt_w1<<<(8 * 16 * 32 * 16) / 256, 256, 0, stream>>>(W1, W1B);
    gcn_cvt_w2<<<(3 * 4 * 32 * 16 + 255) / 256, 256, 0, stream>>>(W2, W2B);

    gcn_gemm1<<<NODES / 16, 256, 0, stream>>>(x, W1B, H1);
    gcn_bias_init<<<(NODES * NH + 255) / 256, 256, 0, stream>>>(AG1, b1, NODES * NH, NH);
    gcn_scatter1<<<(E + 7) / 8, 256, 0, stream>>>(H1, src, dst, ew, AG1, E);
    gcn_relu_swz<<<(NODES * (NH / 2) + 255) / 256, 256, 0, stream>>>(AG1, HBSW);
    gcn_gemm2<<<NODES / 16, 96, 0, stream>>>(HBSW, W2B, T2);
    gcn_bias_init<<<(NODES * NC + 255) / 256, 256, 0, stream>>>(out, b2, NODES * NC, NC);
    gcn_scatter2<<<(E + 7) / 8, 256, 0, stream>>>(T2, src, dst, ew, out, E);
    gcn_log_softmax<<<(NODES + 255) / 256, 256, 0, stream>>>(out, NODES);
}
